// ScaledDotProductAttention_38646115729511
// MI455X (gfx1250) — matrix operand order repaired and checked by reading, not compiled, not run on hardware
//
#include <hip/hip_runtime.h>
#include <hip/hip_bf16.h>

typedef _Float16 v16h __attribute__((ext_vector_type(16)));
typedef float    v8f  __attribute__((ext_vector_type(8)));

#define S_LEN 2048
#define DH    64
#define NBH   32          // B*H
#define CB    32          // key-columns per workgroup (kernel 1)
#define SP    36          // score-slab pitch in floats (8*36 % 64 == 32 -> conflict-free halves)
#define KP    36          // packed-K pitch (uints)
#define VP    68          // packed-V pitch (uints) (8*68 % 64 == 32)
#define BLOCK 256
#define NWAVE 8

__device__ __forceinline__ unsigned pack2h(float x, float y) {
    union { _Float16 h[2]; unsigned u; } p;
    p.h[0] = (_Float16)x; p.h[1] = (_Float16)y;
    return p.u;
}

// ---------------------------------------------------------------------------
// Kernel 1: per (b,h) x 32-column slab: S = QK^T/8 (WMMA f16->f32),
// column softmax (axis = query) entirely in the 320KB LDS, write attention.
// ---------------------------------------------------------------------------
extern "C" __global__ __launch_bounds__(BLOCK)
void attn_scores_colsm(const float* __restrict__ Q, const float* __restrict__ K,
                       const unsigned char* __restrict__ mask,
                       float* __restrict__ attn)
{
    extern __shared__ float smem1[];
    float*    Ssh = smem1;                              // [S_LEN][SP] score slab
    unsigned* Kpk = (unsigned*)(smem1 + S_LEN * SP);    // [DH/2][KP] packed f16 K-block (B layout)
    float*    red = (float*)(Kpk + (DH/2) * KP);        // [2][256] cross-wave reductions

    const int bh   = blockIdx.x;        // 0..31
    const int k0   = blockIdx.y * CB;   // column-block base
    const int tid  = threadIdx.x;
    const int wave = tid >> 5;
    const int lane = tid & 31;
    const int lo   = lane & 15;
    const int hi   = lane >> 4;

    const float* Qb = Q + (size_t)bh * S_LEN * DH;
    const float* Kb = K + (size_t)bh * S_LEN * DH;

    // ---- stage K block, packed f16 pairs over d: Kpk[d2][c] = (K[k0+c][2d2], K[k0+c][2d2+1])
    for (int i = tid; i < CB * (DH / 2); i += BLOCK) {
        int c  = i >> 5;      // 0..31
        int d2 = i & 31;      // 0..31
        const float* kp = Kb + (size_t)(k0 + c) * DH + 2 * d2;
        Kpk[d2 * KP + c] = pack2h(kp[0], kp[1]);
    }
    __syncthreads();

    // ---- compute score slab with v_wmma_f32_16x16x32_f16 (D=64 -> 2 k-steps)
    for (int qt = wave; qt < S_LEN / 16; qt += NWAVE) {
        const int q0 = qt << 4;
        v8f acc0, acc1;
        #pragma unroll
        for (int j = 0; j < 8; ++j) { acc0[j] = 0.0f; acc1[j] = 0.0f; }

        #pragma unroll
        for (int kb = 0; kb < DH; kb += 32) {
            // A fragment: rows q0+lo, K = kb + 16*hi + s (contiguous)
            v16h a;
            const float* qp = Qb + (size_t)(q0 + lo) * DH + kb + 16 * hi;
            #pragma unroll
            for (int t = 0; t < 8; ++t) {
                int k2 = 2 * t;   // VGPR t -> K pair base
                a[2 * t]     = (_Float16)qp[k2];
                a[2 * t + 1] = (_Float16)qp[k2 + 1];
            }
            // B fragments from LDS: VGPR t -> K = kb + 2t + 16*hi (packed pairs)
            v16h b0, b1;
            unsigned* pb0 = (unsigned*)&b0;
            unsigned* pb1 = (unsigned*)&b1;
            const int rbase = (kb >> 1) + 8 * hi;
            #pragma unroll
            for (int t = 0; t < 8; ++t) {
                pb0[t] = Kpk[(rbase + t) * KP + lo];
                pb1[t] = Kpk[(rbase + t) * KP + 16 + lo];
            }
            acc0 = __builtin_amdgcn_wmma_f32_16x16x32_f16(false, a, false, b0,
                                                          (short)0, acc0, false, false);
            acc1 = __builtin_amdgcn_wmma_f32_16x16x32_f16(false, a, false, b1,
                                                          (short)0, acc1, false, false);
        }
        // D layout: VGPR j -> row q0 + j + 8*hi, col = lo (+16). Apply 1/TEMPERATURE.
        #pragma unroll
        for (int j = 0; j < 8; ++j) {
            int r = q0 + j + 8 * hi;
            Ssh[r * SP + lo]      = acc0[j] * 0.125f;
            Ssh[r * SP + 16 + lo] = acc1[j] * 0.125f;
        }
    }
    __syncthreads();

    // ---- column softmax over the query axis (2048 rows). Thread (wave,lane):
    //      column c = lane, row chunk = wave*256 .. +255.
    const int c   = lane;
    const int q0c = wave * (S_LEN / NWAVE);
    const unsigned char* mb = mask + ((size_t)bh * S_LEN) * S_LEN + k0 + c;

    float m = -INFINITY;
    for (int q = q0c; q < q0c + S_LEN / NWAVE; ++q) {
        float s = Ssh[q * SP + c];
        if (mb[(size_t)q * S_LEN] != 0) { s = -INFINITY; Ssh[q * SP + c] = s; }
        m = fmaxf(m, s);
    }
    red[wave * 32 + c] = m;
    __syncthreads();
    float mc = red[c];
    #pragma unroll
    for (int w = 1; w < NWAVE; ++w) mc = fmaxf(mc, red[w * 32 + c]);

    float ssum = 0.0f;
    for (int q = q0c; q < q0c + S_LEN / NWAVE; ++q) {
        float s = Ssh[q * SP + c];
        float e = (s == -INFINITY) ? 0.0f : __expf(s - mc);
        Ssh[q * SP + c] = e;
        ssum += e;
    }
    red[256 + wave * 32 + c] = ssum;
    __syncthreads();
    float tot = red[256 + c];
    #pragma unroll
    for (int w = 1; w < NWAVE; ++w) tot += red[256 + w * 32 + c];
    const float rinv = 1.0f / tot;

    // ---- write normalized attention (32 consecutive k per row, coalesced)
    float* ab = attn + ((size_t)bh * S_LEN) * S_LEN + k0 + c;
    for (int q = q0c; q < q0c + S_LEN / NWAVE; ++q) {
        ab[(size_t)q * S_LEN] = Ssh[q * SP + c] * rinv;
    }
}

// ---------------------------------------------------------------------------
// Kernel 2: out = attention @ V. Whole V (packed f16 K-pairs) staged in LDS;
// each wave streams one 16-row strip of attention through WMMA.
// ---------------------------------------------------------------------------
extern "C" __global__ __launch_bounds__(BLOCK)
void attn_av(const float* __restrict__ attn, const float* __restrict__ V,
             float* __restrict__ out)
{
    extern __shared__ unsigned Vpk[];   // [S_LEN/2][VP]
    const int bh   = blockIdx.x;        // 0..31
    const int sb   = blockIdx.y;        // 0..15 (strip blocks of 8 waves)
    const int tid  = threadIdx.x;
    const int wave = tid >> 5;
    const int lane = tid & 31;
    const int lo   = lane & 15;
    const int hi   = lane >> 4;

    const float* Ab = attn + (size_t)bh * S_LEN * S_LEN;
    const float* Vb = V + (size_t)bh * S_LEN * DH;
    float*       Ob = out + (size_t)bh * S_LEN * DH;

    // ---- stage all of V, packed over K pairs: Vpk[k2][d] = (V[2k2][d], V[2k2+1][d])
    for (int i = tid; i < (S_LEN / 2) * DH; i += BLOCK) {
        int k2 = i >> 6;
        int d  = i & 63;
        Vpk[k2 * VP + d] = pack2h(Vb[(size_t)(2 * k2) * DH + d],
                                  Vb[(size_t)(2 * k2 + 1) * DH + d]);
    }
    __syncthreads();

    const int q0 = ((sb * NWAVE + wave) << 4);   // 16-row output strip
    v8f acc[4];
    #pragma unroll
    for (int nt = 0; nt < 4; ++nt)
        #pragma unroll
        for (int j = 0; j < 8; ++j) acc[nt][j] = 0.0f;

    for (int k = 0; k < S_LEN; k += 32) {
        // A fragment: attention rows, f32 -> f16 on the fly
        v16h a;
        const float* ap = Ab + (size_t)(q0 + lo) * S_LEN + k + 16 * hi;
        #pragma unroll
        for (int t = 0; t < 8; ++t) {
            int k2 = 2 * t;
            a[2 * t]     = (_Float16)ap[k2];
            a[2 * t + 1] = (_Float16)ap[k2 + 1];
        }
        const int rb = (k >> 1) + 8 * hi;
        #pragma unroll
        for (int nt = 0; nt < 4; ++nt) {
            v16h b;
            unsigned* pb = (unsigned*)&b;
            #pragma unroll
            for (int t = 0; t < 8; ++t)
                pb[t] = Vpk[(rb + t) * VP + nt * 16 + lo];
            acc[nt] = __builtin_amdgcn_wmma_f32_16x16x32_f16(false, a, false, b,
                                                             (short)0, acc[nt], false, false);
        }
    }
    #pragma unroll
    for (int nt = 0; nt < 4; ++nt)
        #pragma unroll
        for (int j = 0; j < 8; ++j)
            Ob[(size_t)(q0 + j + 8 * hi) * DH + nt * 16 + lo] = acc[nt][j];
}

// ---------------------------------------------------------------------------
extern "C" void kernel_launch(void* const* d_in, const int* in_sizes, int n_in,
                              void* d_out, int out_size, void* d_ws, size_t ws_size,
                              hipStream_t stream)
{
    const float* Q = (const float*)d_in[0];
    const float* K = (const float*)d_in[1];
    const float* V = (const float*)d_in[2];
    const unsigned char* mask = (const unsigned char*)d_in[3];

    float* out  = (float*)d_out;                          // [B,H,S,D]
    float* attn = out + (size_t)NBH * S_LEN * DH;         // [B,H,S,S]

    // Kernel 1: 32 (b,h) x 64 column blocks; ~301.5 KB dynamic LDS (<= 320 KB/WG)
    size_t lds1 = ((size_t)S_LEN * SP + (size_t)(DH / 2) * KP + 512) * sizeof(float);
    dim3 g1(NBH, S_LEN / CB);
    attn_scores_colsm<<<g1, BLOCK, lds1, stream>>>(Q, K, mask, attn);

    // Kernel 2: 32 (b,h) x 16 strip-blocks; ~272 KB dynamic LDS
    size_t lds2 = (size_t)(S_LEN / 2) * VP * sizeof(unsigned);
    dim3 g2(NBH, (S_LEN / 16) / NWAVE);
    attn_av<<<g2, BLOCK, lds2, stream>>>(attn, V, out);
}
